// PointwiseAttn_80711025426881
// MI455X (gfx1250) — compile-verified, matrix operand order checked by reading, not run on hardware
//
#include <hip/hip_runtime.h>
#include <hip/hip_bf16.h>

// ---------------------------------------------------------------------------
// Pointwise attention, fused, for MI455X (gfx1250, wave32, WMMA + TDM).
//   qh   = (q @ Wq) * 1/sqrt(16)            (WMMA, f16 in / f32 acc)
//   KV_p  --TDM-->  LDS (fp32 64x64 tile, 1MB t-stride, pipelined async)
//   Wkq  = Wk @ Qb(qh)     per pixel        (block-diag query embed)
//   S    = KV @ Wkq        per pixel        (64t x 4g scores)
//   A    = softmax_t(S)                     (in-register, shfl_xor 16)
//   Z    = Ab(A) @ KV ; x = diag_g(Z @ Wv)  (weighted V, no Vh materialized)
//   out  = X16 @ Wout + b_out               (WMMA)
// A-operands read straight from row-major LDS as 2x ds_load_b128 (the ISA
// 7.12.2 A layout is two contiguous 8-element runs per lane). B-operands for
// constant weights are staged pre-packed so fetch is also 2x ds_load_b128.
// ---------------------------------------------------------------------------

typedef __attribute__((ext_vector_type(16))) _Float16 v16h;
typedef __attribute__((ext_vector_type(8)))  _Float16 v8h;
typedef __attribute__((ext_vector_type(8)))  float    v8f;
typedef __attribute__((ext_vector_type(2)))  _Float16 h2;
typedef __attribute__((ext_vector_type(4)))  unsigned u32x4;
typedef __attribute__((ext_vector_type(8)))  int      i32x8;
typedef __attribute__((ext_vector_type(4)))  int      i32x4;

#define NIMG 4
#define TDIM 64
#define LDIM 64
#define QCH  256
#define KVCH 64
#define PIX  (NIMG * LDIM * LDIM)               // 16384 flat pixels
#define KV_TSTRIDE ((size_t)LDIM * LDIM * KVCH) // 262144 floats between t rows

#if defined(__has_builtin)
#if __has_builtin(__builtin_amdgcn_tensor_load_to_lds) && \
    __has_builtin(__builtin_amdgcn_s_wait_tensorcnt)
#define HAVE_TDM 1
#endif
#endif

__device__ __forceinline__ v8f wmma16(v16h a, v16h b, v8f c) {
  return __builtin_amdgcn_wmma_f32_16x16x32_f16(false, a, false, b, (short)0, c,
                                                false, false);
}

// K base (even) for VGPR-pair v (0..7), 16-bit A 16x32 layout (ISA 7.12.2).
__device__ __forceinline__ int kmap(int v, int lane) {
  int b = (v < 4) ? (v * 2) : (16 + (v - 4) * 2);
  return b + ((lane & 16) ? 8 : 0);
}

// Operand-packed offset inside a K=64 buffer: element with non-K index p
// (0..15) and K index k lands in the owning lane at its kmap slot.
__device__ __forceinline__ int pk_off(int p, int k) {
  int kk = k & 31;
  int lh = (kk >> 3) & 1;
  int kr = kk - 8 * lh;
  int slot = (kr & 7) + ((kr >> 4) << 3);
  return (k >> 5) * 512 + (p + 16 * lh) * 16 + slot;
}

// Packed position for B element (k, n) of a weight with ktiles K-chunks.
__device__ __forceinline__ int bpk_idx(int k, int n, int ktiles) {
  int kk = k & 31;
  int lh = (kk >> 3) & 1;
  int kr = kk - 8 * lh;
  int slot = (kr & 7) + ((kr >> 4) << 3);
  return ((n >> 4) * ktiles + (k >> 5)) * 512 + ((n & 15) + 16 * lh) * 16 + slot;
}

// Packed operand fetch: 16 contiguous halves per lane -> 2x ds_load_b128.
__device__ __forceinline__ v16h readPacked(const _Float16* buf, int tile,
                                           int lane) {
  const v8h* p = reinterpret_cast<const v8h*>(buf + tile * 512 + lane * 16);
  v8h lo = p[0], hi = p[1];
  return __builtin_shufflevector(lo, hi, 0, 1, 2, 3, 4, 5, 6, 7, 8, 9, 10, 11,
                                 12, 13, 14, 15);
}

// A operand from row-major f16 LDS: two contiguous 8-half runs per lane.
__device__ __forceinline__ v16h readA(const _Float16* buf, int stride, int m0,
                                      int k0, int lane) {
  const _Float16* row =
      buf + (m0 + (lane & 15)) * stride + k0 + ((lane & 16) ? 8 : 0);
  v8h lo = *reinterpret_cast<const v8h*>(row);       // k: base+0..7
  v8h hi = *reinterpret_cast<const v8h*>(row + 16);  // k: base+16..23
  return __builtin_shufflevector(lo, hi, 0, 1, 2, 3, 4, 5, 6, 7, 8, 9, 10, 11,
                                 12, 13, 14, 15);
}

// A operand from row-major fp32 LDS (TDM tile): 4x ds_load_b128 + cvt.
__device__ __forceinline__ v16h readA_f32(const float* buf, int stride, int m0,
                                          int k0, int lane) {
  const float* row =
      buf + (m0 + (lane & 15)) * stride + k0 + ((lane & 16) ? 8 : 0);
  float4 a0 = *reinterpret_cast<const float4*>(row);
  float4 a1 = *reinterpret_cast<const float4*>(row + 4);
  float4 b0 = *reinterpret_cast<const float4*>(row + 16);
  float4 b1 = *reinterpret_cast<const float4*>(row + 20);
  v16h r;
  r[0] = (_Float16)a0.x;  r[1] = (_Float16)a0.y;
  r[2] = (_Float16)a0.z;  r[3] = (_Float16)a0.w;
  r[4] = (_Float16)a1.x;  r[5] = (_Float16)a1.y;
  r[6] = (_Float16)a1.z;  r[7] = (_Float16)a1.w;
  r[8] = (_Float16)b0.x;  r[9] = (_Float16)b0.y;
  r[10] = (_Float16)b0.z; r[11] = (_Float16)b0.w;
  r[12] = (_Float16)b1.x; r[13] = (_Float16)b1.y;
  r[14] = (_Float16)b1.z; r[15] = (_Float16)b1.w;
  return r;
}

// B operand from row-major fp32 LDS (TDM tile): column gather + f16 convert.
__device__ __forceinline__ v16h readB_f32(const float* buf, int stride, int k0,
                                          int n0, int lane) {
  int n = n0 + (lane & 15);
  v16h r;
#pragma unroll
  for (int v = 0; v < 8; ++v) {
    int k = k0 + kmap(v, lane);
    r[2 * v] = (_Float16)buf[k * stride + n];
    r[2 * v + 1] = (_Float16)buf[(k + 1) * stride + n];
  }
  return r;
}

// Qb[k, n] = (n == k/16) ? qh_p[k] : 0  -- block-diagonal query embedding.
__device__ __forceinline__ v16h readB_qb(const _Float16* qh_p, int k0, int lane) {
  int n = lane & 15;
  v16h r;
#pragma unroll
  for (int v = 0; v < 8; ++v) {
    int k = k0 + kmap(v, lane);
    h2 p = *reinterpret_cast<const h2*>(qh_p + k);
    r[2 * v] = (n == (k >> 4)) ? p.x : (_Float16)0.0f;
    r[2 * v + 1] = (n == ((k + 1) >> 4)) ? p.y : (_Float16)0.0f;
  }
  return r;
}

// Store a 16x16 f32 D tile into a row-major f16 buffer (D layout, ISA 7.12.2).
__device__ __forceinline__ void storeD_h(_Float16* buf, int stride, int m0,
                                         int n0, v8f acc, int lane, float scale) {
  int n = n0 + (lane & 15);
  int mb = m0 + ((lane & 16) ? 8 : 0);
#pragma unroll
  for (int v = 0; v < 8; ++v)
    buf[(mb + v) * stride + n] = (_Float16)(acc[v] * scale);
}

// D tile -> B-operand-packed (next GEMM's K = this tile's row index m).
__device__ __forceinline__ void storeD_Bpacked(_Float16* buf, v8f acc, int m0,
                                               int lane) {
  int n = lane & 15;
  int mb = m0 + ((lane & 16) ? 8 : 0);
#pragma unroll
  for (int v = 0; v < 8; ++v) buf[pk_off(n, mb + v)] = (_Float16)acc[v];
}

// D tile -> A-operand-packed (next GEMM's K = this tile's column index n).
__device__ __forceinline__ void storeD_Apacked(_Float16* buf, v8f acc, int n0,
                                               int lane) {
  int k = n0 + (lane & 15);
  int mb = (lane & 16) ? 8 : 0;
#pragma unroll
  for (int v = 0; v < 8; ++v) buf[pk_off(mb + v, k)] = (_Float16)acc[v];
}

// Issue the KV tile copy: TDM (64x64 fp32 tile, row stride 1MB) -> LDS, or a
// synchronous vector-load fallback when the TDM builtin is unavailable.
__device__ __forceinline__ void kv_issue(float* ldsdst, const float* gsrc,
                                         int lane) {
#if defined(HAVE_TDM)
  unsigned long long ga = (unsigned long long)(uintptr_t)gsrc;
  u32x4 g0;
  g0[0] = 1u;                                       // count=1, user descriptor
  g0[1] = (unsigned)(uintptr_t)ldsdst;              // lds_addr (bytes)
  g0[2] = (unsigned)(ga & 0xffffffffu);             // global_addr lo
  g0[3] = (unsigned)((ga >> 32) & 0x01ffffffu) | (2u << 30);  // hi | type=2
  i32x8 g1;
  g1[0] = (int)(2u << 16);   // data_size=4B; no multicast/iterate/pad
  g1[1] = (int)(64u << 16);  // tensor_dim0 = 64 elems (lo16)
  g1[2] = (int)(64u << 16);  // tensor_dim0 hi=0 | tensor_dim1 = 64 (lo16)
  g1[3] = (int)(64u << 16);  // tensor_dim1 hi=0 | tile_dim0 = 64
  g1[4] = 64;                // tile_dim1 = 64, tile_dim2 = 0
  g1[5] = (int)KV_TSTRIDE;   // tensor_dim0_stride = 262144 (lo32)
  g1[6] = 0;
  g1[7] = 0;
  i32x4 gz = {0, 0, 0, 0};   // 2D tensor: groups 2/3 unused
#if defined(__clang_major__) && __clang_major__ >= 23
  i32x8 gz8 = {0, 0, 0, 0, 0, 0, 0, 0};
  __builtin_amdgcn_tensor_load_to_lds(g0, g1, gz, gz, gz8, 0);
#else
  __builtin_amdgcn_tensor_load_to_lds(g0, g1, gz, gz, 0);
#endif
#else
#pragma unroll 4
  for (int it = 0; it < 32; ++it) {
    int fl = it * 32 + lane;
    int t = fl >> 4;
    int c4 = (fl & 15) * 4;
    float4 v4 = *reinterpret_cast<const float4*>(gsrc + (size_t)t * KV_TSTRIDE + c4);
    *reinterpret_cast<float4*>(ldsdst + t * KVCH + c4) = v4;
  }
#endif
}

__device__ __forceinline__ void kv_wait() {
#if defined(HAVE_TDM)
  __builtin_amdgcn_s_wait_tensorcnt(0);
#endif
  asm volatile("" ::: "memory");
}

__global__ __launch_bounds__(64) void pw_attn_kernel(
    const float* __restrict__ q, const float* __restrict__ kv,
    const float* __restrict__ Wq, const float* __restrict__ Wk,
    const float* __restrict__ Wv, const float* __restrict__ Wout,
    const float* __restrict__ b_out, float* __restrict__ out) {
  // Phase-aliased LDS pool (40KB) + persistent buffers. Total 60KB.
  __shared__ __align__(16) unsigned char s_pool[40960];
  __shared__ _Float16 s_wk[KVCH * KVCH];  // 8KB  Wk f16, row-major (A reads)
  __shared__ _Float16 s_wv[KVCH * KVCH];  // 8KB  Wv f16, B-packed (8 tiles)
  __shared__ _Float16 s_qh[16 * KVCH];    // 2KB  qh (pre-scaled) per tile
  __shared__ _Float16 s_x[16 * KVCH];     // 2KB  attended features X16

  const int tid = threadIdx.x;
  const int wave = tid >> 5;
  const int lane = tid & 31;

  const int base = blockIdx.x * 16;  // flat pixel base (n*4096 + i*64 + j)
  const int n_img = base >> 12;
  const int pij0 = base & 4095;
  const float* kv_n = kv + (size_t)n_img * TDIM * KV_TSTRIDE;

  // ---- Phase 0/1 staging -> f16 LDS. B weights go in packed layout. ------
  _Float16* s_wq = (_Float16*)s_pool;             // 256x64 B-packed (32KB)
  _Float16* s_qbuf = (_Float16*)(s_pool + 32768); // 16x256 row-major (8KB)
  for (int idx = tid; idx < KVCH * KVCH; idx += 64) {
    int k = idx >> 6, n = idx & 63;
    s_wk[idx] = (_Float16)Wk[idx];                    // row-major (A operand)
    s_wv[bpk_idx(k, n, 2)] = (_Float16)Wv[idx];       // B-packed, tile n4*2+kc
  }
  for (int idx = tid; idx < QCH * KVCH; idx += 64) {
    int k = idx >> 6, n = idx & 63;
    s_wq[bpk_idx(k, n, 8)] = (_Float16)Wq[idx];       // B-packed, tile n4*8+kc
  }
  for (int idx = tid; idx < 16 * QCH; idx += 64) {
    int m = idx >> 8, k = idx & 255;
    s_qbuf[idx] = (_Float16)q[((size_t)(base + m)) * QCH + k];
  }
  __syncthreads();

  // ---- Phase 1: qh = (q16 @ Wq) * 0.25, kept f16 in LDS (16x64) ----------
  {
    v8f acc0 = {}, acc1 = {};
    const int nt0 = 2 * wave, nt1 = 2 * wave + 1;
#pragma unroll
    for (int kc = 0; kc < 8; ++kc) {
      v16h a = readA(s_qbuf, QCH, 0, kc * 32, lane);
      v16h b0 = readPacked(s_wq, nt0 * 8 + kc, lane);
      v16h b1 = readPacked(s_wq, nt1 * 8 + kc, lane);
      acc0 = wmma16(a, b0, acc0);
      acc1 = wmma16(a, b1, acc1);
    }
    storeD_h(s_qh, KVCH, 0, nt0 * 16, acc0, lane, 0.25f);  // fold 1/sqrt(16)
    storeD_h(s_qh, KVCH, 0, nt1 * 16, acc1, lane, 0.25f);
  }
  __syncthreads();  // pool (Wq/qbuf) dead after this point

  // ---- Phase 2: decoupled per-wave pipelines (no intra-phase barriers) ---
  unsigned char* wbase = s_pool + wave * 20480;
  float* kvraw = (float*)wbase;                // 64t x 64c fp32 (16KB, TDM)
  _Float16* wz = (_Float16*)(wbase + 16384);   // packed Wkq / Z (2KB, shared)
  _Float16* abuf = (_Float16*)(wbase + 18432); // packed attn A-tiles (2KB)

  // Loop-invariant WMMA operands: Wk A-tiles, Wv B-tiles (16 x v16h regs).
  v16h wk_a[8], wv_b[8];
#pragma unroll
  for (int kc = 0; kc < 2; ++kc)
#pragma unroll
    for (int t4 = 0; t4 < 4; ++t4) {
      wk_a[kc * 4 + t4] = readA(s_wk, KVCH, t4 * 16, kc * 32, lane);
      wv_b[kc * 4 + t4] = readPacked(s_wv, t4 * 2 + kc, lane);
    }

  // Zero attn rows m>=4 once: those packed cells are never rewritten.
  {
    v8h z = {};
    *(v8h*)(abuf + lane * 16) = z;
    *(v8h*)(abuf + lane * 16 + 8) = z;
    *(v8h*)(abuf + 512 + lane * 16) = z;
    *(v8h*)(abuf + 512 + lane * 16 + 8) = z;
  }

  // Prologue: start DMA of this wave's first pixel.
  kv_issue(kvraw, kv_n + (size_t)(pij0 + wave * 8) * KVCH, lane);

  for (int i = 0; i < 8; ++i) {
    const int p = wave * 8 + i;  // pixel within tile

    // b) Wkq = Wk @ Qb(qh_p)  -- overlaps the in-flight TDM transfer.
    {
      v8f acc[4] = {};
#pragma unroll
      for (int kc = 0; kc < 2; ++kc) {
        v16h b = readB_qb(&s_qh[p * KVCH], kc * 32, lane);
#pragma unroll
        for (int mt = 0; mt < 4; ++mt)
          acc[mt] = wmma16(wk_a[kc * 4 + mt], b, acc[mt]);
      }
#pragma unroll
      for (int mt = 0; mt < 4; ++mt) storeD_Bpacked(wz, acc[mt], mt * 16, lane);
    }

    kv_wait();  // s_wait_tensorcnt 0: KV tile resident in LDS

    // c) S = KV @ Wkq  (64t x 16, cols 0..3 = groups)
    v8f sacc[4] = {};
#pragma unroll
    for (int kc = 0; kc < 2; ++kc) {
      v16h b = readPacked(wz, kc, lane);
#pragma unroll
      for (int mt = 0; mt < 4; ++mt) {
        v16h a = readA_f32(kvraw, KVCH, mt * 16, kc * 32, lane);
        sacc[mt] = wmma16(a, b, sacc[mt]);
      }
    }

    // d) softmax over t per group, in registers: column g lives on lanes
    // g and g+16 (D layout), so one shfl_xor(16) completes each reduction.
    {
      float mx = -1e30f;
#pragma unroll
      for (int mt = 0; mt < 4; ++mt)
#pragma unroll
        for (int v = 0; v < 8; ++v) mx = fmaxf(mx, sacc[mt][v]);
      mx = fmaxf(mx, __shfl_xor(mx, 16));
      float sum = 0.0f;
#pragma unroll
      for (int mt = 0; mt < 4; ++mt)
#pragma unroll
        for (int v = 0; v < 8; ++v) {
          float e = __expf(sacc[mt][v] - mx);
          sacc[mt][v] = e;
          sum += e;
        }
      sum += __shfl_xor(sum, 16);
      float inv = 1.0f / sum;
      int g = lane & 15;
      int mb = (lane & 16) ? 8 : 0;
      if (g < 4) {
#pragma unroll
        for (int mt = 0; mt < 4; ++mt)
#pragma unroll
          for (int v = 0; v < 8; ++v)
            abuf[pk_off(g, mt * 16 + mb + v)] = (_Float16)(sacc[mt][v] * inv);
      }
    }

    // e) Z = Ab @ KV  (16 x 64, rows 0..3 = groups); wz reused for Z.
    {
      v8f acc[4] = {};
#pragma unroll
      for (int kc = 0; kc < 2; ++kc) {
        v16h a = readPacked(abuf, kc, lane);
#pragma unroll
        for (int nt = 0; nt < 4; ++nt) {
          v16h b = readB_f32(kvraw, KVCH, kc * 32, nt * 16, lane);
          acc[nt] = wmma16(a, b, acc[nt]);
        }
      }
#pragma unroll
      for (int nt = 0; nt < 4; ++nt) storeD_Apacked(wz, acc[nt], nt * 16, lane);
    }

    // Drain kvraw reads, then start the next pixel's DMA (overlaps f + b).
    if (i < 7) {
      asm volatile("s_wait_dscnt 0x0" ::: "memory");
      kv_issue(kvraw, kv_n + (size_t)(pij0 + p + 1) * KVCH, lane);
    }

    // f) Y = Z @ Wv ; x[g*16+c] = Y[g, g*16+c]  (group-diagonal rows)
    {
      v8f acc[4] = {};
#pragma unroll
      for (int kc = 0; kc < 2; ++kc) {
        v16h a = readPacked(wz, kc, lane);
#pragma unroll
        for (int nt = 0; nt < 4; ++nt)
          acc[nt] = wmma16(a, wv_b[kc * 4 + nt], acc[nt]);
      }
      if (lane < 16) {  // row m = nt lives in VGPR nt on lanes 0..15
        s_x[p * KVCH + 0 * 16 + lane] = (_Float16)acc[0][0];
        s_x[p * KVCH + 1 * 16 + lane] = (_Float16)acc[1][1];
        s_x[p * KVCH + 2 * 16 + lane] = (_Float16)acc[2][2];
        s_x[p * KVCH + 3 * 16 + lane] = (_Float16)acc[3][3];
      }
    }
  }
  __syncthreads();  // both waves done with pool scratch and s_x

  // ---- Phase 3: out = X16 @ Wout + b_out  (16 x 256) ---------------------
  _Float16* s_wout = (_Float16*)s_pool;  // 64x256 B-packed (32KB), pool reuse
  for (int idx = tid; idx < KVCH * QCH; idx += 64) {
    int k = idx >> 8, n = idx & 255;
    s_wout[bpk_idx(k, n, 2)] = (_Float16)Wout[idx];  // tile = (n>>4)*2 + kc
  }
  __syncthreads();
  {
    v8f acc[8] = {};
#pragma unroll
    for (int kc = 0; kc < 2; ++kc) {
      v16h a = readA(s_x, KVCH, 0, kc * 32, lane);
#pragma unroll
      for (int j = 0; j < 8; ++j) {
        int nt = wave * 8 + j;
        v16h b = readPacked(s_wout, nt * 2 + kc, lane);
        acc[j] = wmma16(a, b, acc[j]);
      }
    }
    int mb = (lane & 16) ? 8 : 0;
#pragma unroll
    for (int j = 0; j < 8; ++j) {
      int col = (wave * 8 + j) * 16 + (lane & 15);
      float bia = b_out[col];
#pragma unroll
      for (int v = 0; v < 8; ++v)
        out[((size_t)(base + mb + v)) * QCH + col] = acc[j][v] + bia;
    }
  }
}

extern "C" void kernel_launch(void* const* d_in, const int* in_sizes, int n_in,
                              void* d_out, int out_size, void* d_ws,
                              size_t ws_size, hipStream_t stream) {
  const float* q = (const float*)d_in[0];
  const float* kv = (const float*)d_in[1];
  const float* Wq = (const float*)d_in[2];
  const float* Wk = (const float*)d_in[3];
  const float* Wv = (const float*)d_in[4];
  const float* Wout = (const float*)d_in[5];
  const float* b_out = (const float*)d_in[6];
  float* out = (float*)d_out;

  dim3 grid(PIX / 16);  // 1024 workgroups, 16 pixels each
  dim3 block(64);       // 2 wave32s
  pw_attn_kernel<<<grid, block, 0, stream>>>(q, kv, Wq, Wk, Wv, Wout, b_out,
                                             out);
}
